// TIPSv2VisionAttention_31044023615544
// MI455X (gfx1250) — matrix operand order repaired and checked by reading, not compiled, not run on hardware
//
#include <hip/hip_runtime.h>

// ---------------------------------------------------------------------------
// MI455X (gfx1250) TIPSv2 vision attention.
// Compute-bound (~77 GFLOP vs ~60MB) -> all matmuls via v_wmma_f32_16x16x32_bf16,
// flash-attention softmax (no 1370x1370 tensor in memory), and CDNA5 data
// movement: every tile (GEMM A/B, Q, K, V) streamed into LDS by the Tensor
// Data Mover (tensor_load_to_lds + s_wait_tensorcnt), double-buffered.
// ---------------------------------------------------------------------------

typedef __bf16 v16bf __attribute__((ext_vector_type(16)));
typedef __bf16 v8bf  __attribute__((ext_vector_type(8)));
typedef float  v8f   __attribute__((ext_vector_type(8)));

#define DEV __device__ __forceinline__

#if defined(__AMDGCN__) && defined(__has_builtin)
#  if __has_builtin(__builtin_amdgcn_tensor_load_to_lds)
#    define HAVE_TDM 1
#  endif
#endif
#ifndef HAVE_TDM
#  define HAVE_TDM 0
#endif

static constexpr int BATCH   = 4;
static constexpr int NSEQ    = 1370;
static constexpr int DIM     = 1024;
static constexpr int NHEADS  = 16;
static constexpr int HDIM    = 64;
static constexpr int MROWS   = BATCH * NSEQ;                 // 5480
static constexpr int PLANE   = BATCH * NHEADS * NSEQ * HDIM; // Q/K/V plane elems
static constexpr float SCALE = 0.125f;                       // 64^-0.5

DEV int imin(int a, int b) { return a < b ? a : b; }

DEV __bf16 to_bf16(float f) {
    unsigned u = __builtin_bit_cast(unsigned, f);
    u += 0x7FFFu + ((u >> 16) & 1u);     // round-to-nearest-even
    unsigned short h = (unsigned short)(u >> 16);
    return __builtin_bit_cast(__bf16, h);
}

DEV v16bf cat16(v8bf a, v8bf b) {
    return __builtin_shufflevector(a, b, 0,1,2,3,4,5,6,7,8,9,10,11,12,13,14,15);
}

DEV v8f wmma_bf16(v16bf a, v16bf b, v8f c) {
    return __builtin_amdgcn_wmma_f32_16x16x32_bf16(false, a, false, b, (short)0, c,
                                                   false, false);
}

// ---------------------------------------------------------------------------
// Tensor Data Mover: 2D tile load, tile_d0 x tile_d1 bf16 elements, packed
// row-major into LDS. Rows/cols beyond tensor_d0/tensor_d1 are zero-filled.
// D# layout per CDNA5 ISA; 6-arg builtin form
// (groups 0-3 + unused VADDR4 group + cpol).
// ---------------------------------------------------------------------------
#if HAVE_TDM
typedef unsigned int u32x4 __attribute__((ext_vector_type(4)));
typedef int          i32x8 __attribute__((ext_vector_type(8)));
typedef int          i32x4 __attribute__((ext_vector_type(4)));

DEV void tdm_load_2d(void* lds, const void* gaddr,
                     unsigned tensor_d0, unsigned tensor_d1,
                     unsigned tile_d0, unsigned tile_d1,
                     unsigned long long stride0_elems) {
    unsigned long long ga = (unsigned long long)(size_t)gaddr;
    unsigned lds_off     = (unsigned)(size_t)lds;   // low 32 bits = LDS offset
    u32x4 g0;
    g0[0] = 1u;                                     // count=1 (valid), flags=0
    g0[1] = lds_off;                                // lds_addr (bytes)
    g0[2] = (unsigned)(ga & 0xFFFFFFFFull);         // global_addr[31:0]
    g0[3] = (unsigned)((ga >> 32) & 0x1FFFFFFull)   // global_addr[56:32]
            | (2u << 30);                           // type=2 ("image")
    i32x8 g1;
    g1[0] = 0x00010000;                             // data_size=1 (2 bytes)
    g1[1] = (int)((tensor_d0 & 0xFFFFu) << 16);     // tensor_dim0[15:0]
    g1[2] = (int)(((tensor_d0 >> 16) & 0xFFFFu) | ((tensor_d1 & 0xFFFFu) << 16));
    g1[3] = (int)(((tensor_d1 >> 16) & 0xFFFFu) | (tile_d0 << 16));  // tile_dim0
    g1[4] = (int)(tile_d1 & 0xFFFFu);               // tile_dim1, tile_dim2=0
    g1[5] = (int)(unsigned)(stride0_elems & 0xFFFFFFFFull);          // stride0 lo
    g1[6] = (int)(unsigned)((stride0_elems >> 32) & 0xFFFFull);      // stride1=0
    g1[7] = 0;
    i32x4 g2; g2[0] = 1; g2[1] = 1; g2[2] = 0; g2[3] = 0;  // dim2=1, dim3=1
    i32x4 g3; g3[0] = 0; g3[1] = 0x00010000; g3[2] = 0; g3[3] = 0;  // dim4=1
    i32x8 g4 = {0, 0, 0, 0, 0, 0, 0, 0};            // unused VADDR4 group
    __builtin_amdgcn_tensor_load_to_lds(g0, g1, g2, g3, g4, 0);
}
#endif

// synchronous 16B copy (fallback staging path)
DEV void cp16(void* lds, const void* g) { *(v8bf*)lds = *(const v8bf*)g; }

// ---------------------------------------------------------------------------
// fp32 -> bf16 conversion (grid-stride)
// ---------------------------------------------------------------------------
__global__ void cvt_f32_bf16_kernel(const float* __restrict__ src,
                                    __bf16* __restrict__ dst, int n) {
    int i = blockIdx.x * blockDim.x + threadIdx.x;
    int stride = gridDim.x * blockDim.x;
    for (; i < n; i += stride) dst[i] = to_bf16(src[i]);
}

// fp32 [R][C] -> bf16 transposed [C][R] (32x32 LDS tiles, R,C multiples of 32)
__global__ __launch_bounds__(256) void cvt_transpose_kernel(
    const float* __restrict__ src, __bf16* __restrict__ dst, int R, int C) {
    __shared__ float tile[32][33];
    int c0 = blockIdx.x * 32, r0 = blockIdx.y * 32;
    int tx = threadIdx.x & 31, ty = threadIdx.x >> 5;   // 32 x 8
#pragma unroll
    for (int i = 0; i < 32; i += 8)
        tile[ty + i][tx] = src[(size_t)(r0 + ty + i) * C + c0 + tx];
    __syncthreads();
#pragma unroll
    for (int i = 0; i < 32; i += 8)
        dst[(size_t)(c0 + ty + i) * R + r0 + tx] = to_bf16(tile[tx][ty + i]);
}

// ---------------------------------------------------------------------------
// Tiled bf16 GEMM with TDM-streamed, double-buffered LDS tiles.
// X:[M][K] bf16 row-major, Wt:[N][K] bf16 (pre-transposed weights).
// 128x128x32 block, 8 waves, wave tile 32x64 (2x4 WMMA accumulators).
// MODE 0: +bias, scatter Q(*scale)/K planes [b,h,n,64], V plane [b,h,64,n]
// MODE 1: +bias, f32 out [5480][1024]
// ---------------------------------------------------------------------------
template <int MODE>
__global__ __launch_bounds__(256) void gemm_kernel(
    const __bf16* __restrict__ X, const __bf16* __restrict__ Wt,
    const float* __restrict__ bias, __bf16* __restrict__ qkv_out,
    float* __restrict__ f32_out) {
    constexpr int N  = (MODE == 0) ? 3 * DIM : DIM;
    constexpr int K  = DIM;
    constexpr int M  = MROWS;
    constexpr int SA = 32;   // packed LDS strides (TDM packs tiles contiguously)

    __shared__ __align__(16) __bf16 As[2][128 * SA];   // [row][k]
    __shared__ __align__(16) __bf16 Bs[2][128 * SA];   // [n][k]

    const int tid  = threadIdx.x;
    const int lane = tid & 31;
    const int wave = tid >> 5;
    const int hi   = lane >> 4;
    const int ln   = lane & 15;
    const int wm   = wave & 3;
    const int wn   = wave >> 2;
    const int m0   = blockIdx.x * 128;
    const int n0   = blockIdx.y * 128;

    v8f acc[2][4];
#pragma unroll
    for (int sm = 0; sm < 2; sm++)
#pragma unroll
        for (int sn = 0; sn < 4; sn++)
#pragma unroll
            for (int g = 0; g < 8; g++) acc[sm][sn][g] = 0.0f;

    auto stage = [&](int buf, int k0) {
#if HAVE_TDM
        if (wave == 0) {
            tdm_load_2d(As[buf], X + (size_t)m0 * K + k0, K - k0, M - m0, 32, 128, K);
            tdm_load_2d(Bs[buf], Wt + (size_t)n0 * K + k0, K - k0, N - n0, 32, 128, K);
        }
#else
#pragma unroll
        for (int it = 0; it < 2; it++) {
            int c = tid + it * 256;            // 512 chunks of 8 bf16
            int row = c >> 2, kc = (c & 3) * 8;
            int gr = imin(m0 + row, M - 1);
            cp16(As[buf] + row * SA + kc, X + (size_t)gr * K + k0 + kc);
            cp16(Bs[buf] + row * SA + kc, Wt + (size_t)(n0 + row) * K + k0 + kc);
        }
#endif
    };

    stage(0, 0);
    for (int kt = 0; kt < K / 32; kt++) {
        const int cur = kt & 1;
        const bool has_next = (kt + 1 < K / 32);
        if (has_next) stage(cur ^ 1, (kt + 1) * 32);
#if HAVE_TDM
        if (wave == 0) {
            if (has_next) __builtin_amdgcn_s_wait_tensorcnt(2);
            else          __builtin_amdgcn_s_wait_tensorcnt(0);
        }
#endif
        __syncthreads();

        v16bf af[2];
#pragma unroll
        for (int sm = 0; sm < 2; sm++) {
            const __bf16* base = As[cur] + (wm * 32 + sm * 16 + ln) * SA;
            af[sm] = cat16(*(const v8bf*)(base + 8 * hi),
                           *(const v8bf*)(base + 8 * hi + 16));
        }
        v16bf bf[4];
#pragma unroll
        for (int sn = 0; sn < 4; sn++) {
            const __bf16* base = Bs[cur] + (wn * 64 + sn * 16 + ln) * SA;
            bf[sn] = cat16(*(const v8bf*)(base + 8 * hi), *(const v8bf*)(base + 8 * hi + 16));
        }
#pragma unroll
        for (int sm = 0; sm < 2; sm++)
#pragma unroll
            for (int sn = 0; sn < 4; sn++)
                acc[sm][sn] = wmma_bf16(af[sm], bf[sn], acc[sm][sn]);
        __syncthreads();
    }

#pragma unroll
    for (int sm = 0; sm < 2; sm++)
#pragma unroll
        for (int sn = 0; sn < 4; sn++) {
            int c = n0 + wn * 64 + sn * 16 + ln;
            float bv = bias[c];
#pragma unroll
            for (int g = 0; g < 8; g++) {
                int r = m0 + wm * 32 + sm * 16 + g + 8 * hi;
                if (r >= M) continue;
                float v = acc[sm][sn][g] + bv;
                if (MODE == 0) {
                    int which = c >> 10, rem = c & 1023;
                    int hh = rem >> 6, dd = rem & 63;
                    if (which == 0) v *= SCALE;
                    int bb = r / NSEQ, nn = r - bb * NSEQ;
                    size_t bh = (size_t)(bb * NHEADS + hh);
                    size_t idx = (which == 2)
                                     ? (bh * HDIM + dd) * NSEQ + nn   // V: [b,h,d,n]
                                     : (bh * NSEQ + nn) * HDIM + dd;  // Q,K: [b,h,n,d]
                    qkv_out[(size_t)which * PLANE + idx] = to_bf16(v);
                } else {
                    f32_out[(size_t)r * DIM + c] = v;
                }
            }
        }
}

// ---------------------------------------------------------------------------
// Flash attention: block = 64 Q rows x one (b,h); 4 waves x 16 rows.
// Q (64x64), K chunks [32 kv][64 d] and V chunks [64 d][32 kv] all streamed
// via TDM (zero-filled past n=1370); K/V double-buffered.
// ---------------------------------------------------------------------------
__global__ __launch_bounds__(128) void attn_kernel(const __bf16* __restrict__ qkv,
                                                   __bf16* __restrict__ Ob) {
    constexpr int SQ = 64;   // packed strides (TDM-compatible)
    constexpr int SV = 32;
    constexpr int SP = 40;   // Ps stride
    constexpr int NCH = (NSEQ + 31) / 32;  // 43

    __shared__ __align__(16) __bf16 Qs[64 * SQ];
    __shared__ __align__(16) __bf16 Ks[2][32 * SQ];
    __shared__ __align__(16) __bf16 Vt[2][64 * SV];
    __shared__ __align__(16) __bf16 Ps[4][16 * SP];

    const int tid  = threadIdx.x;
    const int lane = tid & 31;
    const int wave = tid >> 5;
    const int hi   = lane >> 4;
    const int ln   = lane & 15;
    const int bh   = blockIdx.y;
    const int bb   = bh >> 4;
    const int hh   = bh & 15;
    const int q0   = blockIdx.x * 64;

    const __bf16* Qg = qkv + 0 * (size_t)PLANE + (size_t)bh * NSEQ * HDIM; // [n][d]
    const __bf16* Kg = qkv + 1 * (size_t)PLANE + (size_t)bh * NSEQ * HDIM; // [n][d]
    const __bf16* Vg = qkv + 2 * (size_t)PLANE + (size_t)bh * HDIM * NSEQ; // [d][n]

    auto stage_kv = [&](int buf, int kb) {
#if HAVE_TDM
        if (wave == 0) {
            // K: tile 64(d) x 32(kv rows); rows >= NSEQ-kb zero-filled
            tdm_load_2d(Ks[buf], Kg + (size_t)kb * HDIM, HDIM, NSEQ - kb, HDIM, 32, HDIM);
            // V: tile 32(kv) x 64(d rows); cols >= NSEQ-kb zero-filled
            tdm_load_2d(Vt[buf], Vg + kb, NSEQ - kb, HDIM, 32, HDIM, NSEQ);
        }
#else
#pragma unroll
        for (int it = 0; it < 2; it++) {
            int c = tid + it * 128;            // K: 256 chunks of 8
            int row = c >> 3, col = (c & 7) * 8;
            int gr = imin(kb + row, NSEQ - 1);
            cp16(Ks[buf] + row * SQ + col, Kg + (size_t)gr * HDIM + col);
        }
#pragma unroll
        for (int it = 0; it < 2; it++) {
            int c = tid + it * 128;            // V: 256 chunks of 8
            int d = c >> 2, cc = (c & 3) * 8;
            cp16(Vt[buf] + d * SV + cc, Vg + (size_t)d * NSEQ + kb + cc);
        }
#endif
    };

    // stage Q tile (64x64): TDM zero-fills rows >= NSEQ-q0
#if HAVE_TDM
    if (wave == 0)
        tdm_load_2d(Qs, Qg + (size_t)q0 * HDIM, HDIM, NSEQ - q0, HDIM, 64, HDIM);
#else
#pragma unroll
    for (int it = 0; it < 4; it++) {
        int c = tid + it * 128;                // 512 chunks of 8
        int row = c >> 3, col = (c & 7) * 8;
        int gr = imin(q0 + row, NSEQ - 1);
        cp16(Qs + row * SQ + col, Qg + (size_t)gr * HDIM + col);
    }
#endif

    stage_kv(0, 0);
#if HAVE_TDM
    // TDM ops complete in order: waiting to <=2 leaves only the KV0 pair
    if (wave == 0) __builtin_amdgcn_s_wait_tensorcnt(2);
#endif
    __syncthreads();

    v16bf qa[2];
#pragma unroll
    for (int kh = 0; kh < 2; kh++) {
        const __bf16* base = Qs + (wave * 16 + ln) * SQ + kh * 32;
        qa[kh] = cat16(*(const v8bf*)(base + 8 * hi),
                       *(const v8bf*)(base + 8 * hi + 16));
    }

    v8f O[4];
#pragma unroll
    for (int d = 0; d < 4; d++)
#pragma unroll
        for (int g = 0; g < 8; g++) O[d][g] = 0.0f;
    float mrow[8], lrow[8];
#pragma unroll
    for (int g = 0; g < 8; g++) { mrow[g] = -1e30f; lrow[g] = 0.0f; }

    for (int kt = 0; kt < NCH; kt++) {
        const int cur = kt & 1;
        const int kb  = kt * 32;
        const bool has_next = (kt + 1 < NCH);
        if (has_next) stage_kv(cur ^ 1, kb + 32);
#if HAVE_TDM
        if (wave == 0) {
            if (has_next) __builtin_amdgcn_s_wait_tensorcnt(2);
            else          __builtin_amdgcn_s_wait_tensorcnt(0);
        }
#endif
        __syncthreads();

        // S = Q K^T (two 16x16 kv tiles, contracted over d=64)
        v8f s[2];
#pragma unroll
        for (int sf = 0; sf < 2; sf++) {
#pragma unroll
            for (int g = 0; g < 8; g++) s[sf][g] = 0.0f;
#pragma unroll
            for (int kh = 0; kh < 2; kh++) {
                const __bf16* base = Ks[cur] + (sf * 16 + ln) * SQ + kh * 32;
                v16bf kf = cat16(*(const v8bf*)(base + 8 * hi), *(const v8bf*)(base + 8 * hi + 16));
                s[sf] = wmma_bf16(qa[kh], kf, s[sf]);
            }
            int kv = kb + sf * 16 + ln;
            if (kv >= NSEQ) {
#pragma unroll
                for (int g = 0; g < 8; g++) s[sf][g] = -1e30f;
            }
        }

        // online softmax (C-frag row = g + 8*hi; reduce over 16-lane group)
#pragma unroll
        for (int g = 0; g < 8; g++) {
            float v = fmaxf(s[0][g], s[1][g]);
#pragma unroll
            for (int off = 8; off >= 1; off >>= 1)
                v = fmaxf(v, __shfl_xor(v, off, 16));
            float mn    = fmaxf(mrow[g], v);
            float alpha = __expf(mrow[g] - mn);
            mrow[g]     = mn;
            float p0 = __expf(s[0][g] - mn);
            float p1 = __expf(s[1][g] - mn);
            int prow = g + 8 * hi;
            Ps[wave][prow * SP + ln]      = to_bf16(p0);
            Ps[wave][prow * SP + 16 + ln] = to_bf16(p1);
            float ps = p0 + p1;
#pragma unroll
            for (int off = 8; off >= 1; off >>= 1) ps += __shfl_xor(ps, off, 16);
            lrow[g] = lrow[g] * alpha + ps;
#pragma unroll
            for (int d = 0; d < 4; d++) O[d][g] *= alpha;
        }
        asm volatile("s_wait_dscnt 0" ::: "memory");  // Ps stores -> frag loads

        v16bf pa;
        {
            const __bf16* base = Ps[wave] + ln * SP;
            pa = cat16(*(const v8bf*)(base + 8 * hi),
                       *(const v8bf*)(base + 8 * hi + 16));
        }
#pragma unroll
        for (int ds = 0; ds < 4; ds++) {
            const __bf16* base = Vt[cur] + (ds * 16 + ln) * SV;
            v16bf vf = cat16(*(const v8bf*)(base + 8 * hi), *(const v8bf*)(base + 8 * hi + 16));
            O[ds] = wmma_bf16(pa, vf, O[ds]);
        }
        __syncthreads();
    }

#pragma unroll
    for (int g = 0; g < 8; g++) {
        int r = q0 + wave * 16 + g + 8 * hi;
        if (r >= NSEQ) continue;
        float inv = 1.0f / lrow[g];
        size_t orow = (size_t)(bb * NSEQ + r);
#pragma unroll
        for (int ds = 0; ds < 4; ds++) {
            int col = hh * 64 + ds * 16 + ln;
            Ob[orow * DIM + col] = to_bf16(O[ds][g] * inv);
        }
    }
}

// ---------------------------------------------------------------------------
extern "C" void kernel_launch(void* const* d_in, const int* in_sizes, int n_in,
                              void* d_out, int out_size, void* d_ws, size_t ws_size,
                              hipStream_t stream) {
    const float* x      = (const float*)d_in[0];
    const float* w_qkv  = (const float*)d_in[1];
    const float* b_qkv  = (const float*)d_in[2];
    const float* w_proj = (const float*)d_in[3];
    const float* b_proj = (const float*)d_in[4];
    float* out = (float*)d_out;

    __bf16* Xb     = (__bf16*)d_ws;                   // [5480][1024]
    __bf16* Wqkvt  = Xb + (size_t)MROWS * DIM;        // [3072][1024] (W^T)
    __bf16* Wprojt = Wqkvt + (size_t)3 * DIM * DIM;   // [1024][1024] (W^T)
    __bf16* QKVb   = Wprojt + (size_t)DIM * DIM;      // 3 planes
    __bf16* Ob     = QKVb + 3 * (size_t)PLANE;        // [5480][1024]

    cvt_f32_bf16_kernel<<<2048, 256, 0, stream>>>(x, Xb, MROWS * DIM);
    cvt_transpose_kernel<<<dim3(3 * DIM / 32, DIM / 32), 256, 0, stream>>>(
        w_qkv, Wqkvt, DIM, 3 * DIM);
    cvt_transpose_kernel<<<dim3(DIM / 32, DIM / 32), 256, 0, stream>>>(
        w_proj, Wprojt, DIM, DIM);

    gemm_kernel<0><<<dim3((MROWS + 127) / 128, (3 * DIM) / 128), 256, 0, stream>>>(
        Xb, Wqkvt, b_qkv, QKVb, nullptr);

    attn_kernel<<<dim3((NSEQ + 63) / 64, BATCH * NHEADS), 128, 0, stream>>>(QKVb, Ob);

    gemm_kernel<1><<<dim3((MROWS + 127) / 128, DIM / 128), 256, 0, stream>>>(
        Ob, Wprojt, b_proj, nullptr, out);
}
